// Decoder_75007308857772
// MI455X (gfx1250) — matrix operand order repaired and checked by reading, not compiled, not run on hardware
//
#include <hip/hip_runtime.h>
#include <hip/hip_bf16.h>
#include <stdint.h>

#define B_   4
#define S_   1024
#define D_   1024
#define H_   16
#define DK_  64
#define DFF_ 4096
#define TOK  (B_*S_)

typedef __attribute__((ext_vector_type(16))) __bf16 v16bf;
typedef __attribute__((ext_vector_type(8)))  float  v8f;
typedef __attribute__((ext_vector_type(4)))  unsigned int u32x4;
typedef __attribute__((ext_vector_type(8)))  unsigned int u32x8;

union Frag {
  v16bf v;
  uint4 q[2];
};

static __device__ __forceinline__ v8f wmma_bf16(v16bf a, v16bf b, v8f c) {
  // 8 args: (neg_a, A, neg_b, B, c_mod, C, reuse_a, reuse_b)
  return __builtin_amdgcn_wmma_f32_16x16x32_bf16(false, a, false, b, (short)0, c, false, false);
}

// ------------------------------------------------------------------
// TDM: DMA one 128-row x 32-col bf16 tile (row stride = strideElems)
// from global memory into LDS, padding each 16-DWORD row with 4 DWORDs
// so the LDS row stride is 40 halves (80B: 16B-aligned, bank-conflict
// free for the WMMA fragment loads).  D# per CDNA5 ISA.
// ------------------------------------------------------------------
static __device__ __forceinline__ void tdm_load_tile(const __bf16* gptr,
                                                     unsigned lds_off,
                                                     unsigned strideElems)
{
  unsigned long long ga = (unsigned long long)(uintptr_t)gptr;
  u32x4 g0;
  g0[0] = 1u;                                   // count=1, user descriptor
  g0[1] = lds_off;                              // lds_addr (bytes)
  g0[2] = (unsigned)ga;                         // global_addr[31:0]
  g0[3] = (unsigned)(ga >> 32) | (2u << 30);    // global_addr[56:32] | type=2
  u32x8 g1;
  const unsigned td = 0x40000000u;              // huge tensor dims: no OOB
  g1[0] = (1u << 16)      // data_size = 2 bytes
        | (1u << 20)      // pad_enable
        | (3u << 22)      // pad_interval: 16 DWORDs
        | (3u << 25);     // pad_amount:   4 DWORDs
  g1[1] = (td & 0xFFFFu) << 16;                     // tensor_dim0[15:0]
  g1[2] = (td >> 16) | ((td & 0xFFFFu) << 16);      // dim0[31:16] | dim1[15:0]
  g1[3] = (td >> 16) | (32u << 16);                 // dim1[31:16] | tile_dim0=32
  g1[4] = 128u;                                     // tile_dim1=128, tile_dim2=0
  g1[5] = strideElems;                              // tensor_dim0_stride[31:0]
  g1[6] = 0u;                                       // stride hi, dim1_stride lo
  g1[7] = 0u;
  u32x4 gz = {0u, 0u, 0u, 0u};                      // groups 2/3: tile_dim3/4 = 0
  asm volatile("tensor_load_to_lds %0, %1, %2, %3"
               :: "s"(g0), "s"(g1), "s"(gz), "s"(gz)
               : "memory");
}

// ------------------------------------------------------------------
// fp32 -> bf16 conversion
// ------------------------------------------------------------------
__global__ __launch_bounds__(256)
void k_cvt_bf16(const float* __restrict__ in, __bf16* __restrict__ out, int n) {
  int i = blockIdx.x * 256 + threadIdx.x;
  if (i < n) out[i] = (__bf16)in[i];
}

// ------------------------------------------------------------------
// GEMM: C[m,n] = sum_k A[m,k] * W[n,k] + bias[n]  (A: MxK bf16, W: NxK bf16)
// 256 threads = 8 waves, 128x128 tile, K-step 32.
// TDM double-buffered global->LDS, WMMA f32_16x16x32_bf16 compute.
// ------------------------------------------------------------------
#define LSTR 40   // padded LDS row stride in halves (set by TDM pad fields)

__global__ __launch_bounds__(256)
void k_gemm(const __bf16* __restrict__ A, const __bf16* __restrict__ W,
            const float* __restrict__ bias,
            float* __restrict__ outF, __bf16* __restrict__ outB,
            int M, int N, int K, int relu)
{
  __shared__ __bf16 lA[2][128 * LSTR];
  __shared__ __bf16 lB[2][128 * LSTR];

  const int tid  = threadIdx.x;
  const int lane = tid & 31;
  const int wid  = tid >> 5;
  const int hl   = lane >> 4;        // half-wave select
  const int lm   = lane & 15;
  const int wm   = (wid >> 2) * 64;  // wave M offset in tile (0 / 64)
  const int wn   = (wid & 3) * 32;   // wave N offset in tile (0/32/64/96)
  const size_t tM = (size_t)blockIdx.y * 128;
  const size_t tN = (size_t)blockIdx.x * 128;

  const __bf16* gA = A + tM * (size_t)K;
  const __bf16* gW = W + tN * (size_t)K;
  const unsigned la0 = (unsigned)(uintptr_t)&lA[0][0];
  const unsigned la1 = (unsigned)(uintptr_t)&lA[1][0];
  const unsigned lb0 = (unsigned)(uintptr_t)&lB[0][0];
  const unsigned lb1 = (unsigned)(uintptr_t)&lB[1][0];

  v8f acc[4][2] = {};
  const int ntiles = K >> 5;

  // prologue: DMA tile 0 into buffer 0 (one wave drives the TDM)
  if (wid == 0) {
    tdm_load_tile(gA, la0, (unsigned)K);
    tdm_load_tile(gW, lb0, (unsigned)K);
  }

  for (int i = 0; i < ntiles; ++i) {
    if (wid == 0) {
      if (i + 1 < ntiles) {
        // prefetch next tile into the other buffer, then wait for tile i:
        // TENSORcnt is in-order, so <=2 outstanding means tile i landed.
        const unsigned nb = (unsigned)(i + 1) & 1u;
        tdm_load_tile(gA + (size_t)(i + 1) * 32, nb ? la1 : la0, (unsigned)K);
        tdm_load_tile(gW + (size_t)(i + 1) * 32, nb ? lb1 : lb0, (unsigned)K);
        __builtin_amdgcn_s_wait_tensorcnt(2);
      } else {
        __builtin_amdgcn_s_wait_tensorcnt(0);
      }
    }
    __syncthreads();   // tile i visible to all 8 waves

    const __bf16* bufA = lA[i & 1];
    const __bf16* bufB = lB[i & 1];
    Frag af[4], bfv[2];
#pragma unroll
    for (int mf = 0; mf < 4; ++mf) {
      // A frag: lane holds row M=lm; hl=0 -> K{0..7,16..23}, hl=1 -> K{8..15,24..31}
      const __bf16* p = &bufA[(wm + mf * 16 + lm) * LSTR + hl * 8];
      af[mf].q[0] = *(const uint4*)p;
      af[mf].q[1] = *(const uint4*)(p + 16);
    }
#pragma unroll
    for (int nf = 0; nf < 2; ++nf) {
      // B frag: lane holds col N=lm; hl=0 -> K{0..7,16..23}, hl=1 -> K{8..15,24..31}
      const __bf16* p = &bufB[(wn + nf * 16 + lm) * LSTR + hl * 8];
      bfv[nf].q[0] = *(const uint4*)p;
      bfv[nf].q[1] = *(const uint4*)(p + 16);
    }
#pragma unroll
    for (int mf = 0; mf < 4; ++mf)
#pragma unroll
      for (int nf = 0; nf < 2; ++nf)
        acc[mf][nf] = wmma_bf16(af[mf].v, bfv[nf].v, acc[mf][nf]);

    __syncthreads();   // this buffer free before the TDM 2 iters ahead rewrites it
  }

  // epilogue: C/D layout -> VGPR r: (lane<16) M=r, else M=8+r; N=lm
#pragma unroll
  for (int nf = 0; nf < 2; ++nf) {
    size_t ng = tN + wn + nf * 16 + lm;
    float bb = bias[ng];
#pragma unroll
    for (int mf = 0; mf < 4; ++mf) {
#pragma unroll
      for (int r = 0; r < 8; ++r) {
        size_t mg = tM + wm + mf * 16 + hl * 8 + r;
        float v = acc[mf][nf][r] + bb;
        if (relu) v = fmaxf(v, 0.0f);
        if (outF) outF[mg * (size_t)N + ng] = v;
        if (outB) outB[mg * (size_t)N + ng] = (__bf16)v;
      }
    }
  }
}

// ------------------------------------------------------------------
// V transpose: vt[b,h,d,s] = v[b,s,h*DK+d]
// ------------------------------------------------------------------
__global__ __launch_bounds__(256)
void k_transpose_v(const __bf16* __restrict__ v, __bf16* __restrict__ vt) {
  int i = blockIdx.x * 256 + threadIdx.x;   // over B*H*DK*S, s fastest
  int s = i & (S_ - 1);
  int t = i >> 10;
  int d = t & (DK_ - 1);
  t >>= 6;
  int h = t & (H_ - 1);
  int b = t >> 4;
  vt[i] = v[((size_t)b * S_ + s) * D_ + h * DK_ + d];
}

// ------------------------------------------------------------------
// Flash attention: 128 threads = 4 waves, each wave owns 16 q rows.
// Scores and P*V via WMMA; P transposed wave-locally through LDS,
// ordered with the CDNA5 split DS counter (s_wait_dscnt 0).
// ------------------------------------------------------------------
__global__ __launch_bounds__(128)
void k_attn(const __bf16* __restrict__ Q, const __bf16* __restrict__ Kt,
            const __bf16* __restrict__ Vt, __bf16* __restrict__ O, int causal)
{
  __shared__ __bf16 lp[4][16 * 40];

  const int lane = threadIdx.x & 31, w = threadIdx.x >> 5;
  const int hl = lane >> 4, lm = lane & 15;
  const int nqt = S_ / 64;
  const int qt = blockIdx.x % nqt;
  const int h  = (blockIdx.x / nqt) % H_;
  const int b  = blockIdx.x / (nqt * H_);
  const int qbase = qt * 64 + w * 16;

  const __bf16* qp = Q + (size_t)b * S_ * D_ + (size_t)h * DK_;
  const __bf16* kp = Kt + (size_t)b * S_ * D_ + (size_t)h * DK_;
  const __bf16* vp = Vt + ((size_t)b * H_ + h) * (size_t)DK_ * S_;
  __bf16* op = O + (size_t)b * S_ * D_ + (size_t)h * DK_;

  // Q fragments (16 rows x 64 d) -> 2 A-frags, K-dim 32 each
  Frag aq[2];
  const __bf16* qrow = qp + (size_t)(qbase + lm) * D_;
#pragma unroll
  for (int f = 0; f < 2; ++f) {
    aq[f].q[0] = *(const uint4*)(qrow + f * 32 + hl * 8);
    aq[f].q[1] = *(const uint4*)(qrow + f * 32 + hl * 8 + 16);
  }

  v8f o4[4] = {};
  float mi[8], li[8];
#pragma unroll
  for (int r = 0; r < 8; ++r) { mi[r] = -3.0e38f; li[r] = 0.0f; }

  __bf16* lpp = lp[w];
  const int kmax = causal ? (qt * 64 + 64) : S_;

  for (int kc = 0; kc < kmax; kc += 32) {
    // ---- scores: two 16-col fragments over 32 keys ----
    v8f s[2];
#pragma unroll
    for (int nb = 0; nb < 2; ++nb) {
      Frag bk0, bk1;
      const __bf16* krow = kp + (size_t)(kc + nb * 16 + lm) * D_ + hl * 8;
      bk0.q[0] = *(const uint4*)(krow);
      bk0.q[1] = *(const uint4*)(krow + 16);
      bk1.q[0] = *(const uint4*)(krow + 32);
      bk1.q[1] = *(const uint4*)(krow + 48);
      v8f c = {};
      c = wmma_bf16(aq[0].v, bk0.v, c);
      c = wmma_bf16(aq[1].v, bk1.v, c);
      const int kg = kc + nb * 16 + lm;
#pragma unroll
      for (int r = 0; r < 8; ++r) {
        float v = c[r] * 0.125f;              // 1/sqrt(64)
        if (causal && kg > qbase + hl * 8 + r) v = -1.0e9f;
        s[nb][r] = v;
      }
    }

    // ---- online softmax update (rows live in 16-lane half-groups) ----
    float alpha[8];
#pragma unroll
    for (int r = 0; r < 8; ++r) {
      float mx = fmaxf(s[0][r], s[1][r]);
#pragma unroll
      for (int x = 1; x < 16; x <<= 1) mx = fmaxf(mx, __shfl_xor(mx, x, 32));
      float mn = fmaxf(mi[r], mx);
      alpha[r] = __expf(mi[r] - mn);
      float p0 = __expf(s[0][r] - mn);
      float p1 = __expf(s[1][r] - mn);
      s[0][r] = p0; s[1][r] = p1;
      float rs = p0 + p1;
#pragma unroll
      for (int x = 1; x < 16; x <<= 1) rs += __shfl_xor(rs, x, 32);
      li[r] = li[r] * alpha[r] + rs;
      mi[r] = mn;
    }
#pragma unroll
    for (int nb2 = 0; nb2 < 4; ++nb2)
#pragma unroll
      for (int r = 0; r < 8; ++r) o4[nb2][r] *= alpha[r];

    // ---- transpose P (16x32) through wave-private LDS into A-frag layout ----
#pragma unroll
    for (int r = 0; r < 8; ++r) {
      int m = hl * 8 + r;
      lpp[m * 40 + lm]      = (__bf16)s[0][r];
      lpp[m * 40 + 16 + lm] = (__bf16)s[1][r];
    }
    asm volatile("s_wait_dscnt 0" ::: "memory");  // per-wave DS ordering (CDNA5 split counters)
    Frag pa;
    {
      const __bf16* p = &lpp[lm * 40 + hl * 8];
      pa.q[0] = *(const uint4*)p;
      pa.q[1] = *(const uint4*)(p + 16);
    }

    // ---- O += P @ V   (B-frags from pre-transposed Vt: contiguous rows) ----
#pragma unroll
    for (int nb2 = 0; nb2 < 4; ++nb2) {
      Frag bvf;
      const __bf16* vrow = vp + (size_t)(nb2 * 16 + lm) * S_ + kc + hl * 8;
      bvf.q[0] = *(const uint4*)vrow;
      bvf.q[1] = *(const uint4*)(vrow + 16);
      o4[nb2] = wmma_bf16(pa.v, bvf.v, o4[nb2]);
    }
  }

  // ---- normalize + store bf16 ----
#pragma unroll
  for (int nb2 = 0; nb2 < 4; ++nb2) {
#pragma unroll
    for (int r = 0; r < 8; ++r) {
      size_t mg = (size_t)(qbase + hl * 8 + r);
      op[mg * D_ + nb2 * 16 + lm] = (__bf16)(o4[nb2][r] / li[r]);
    }
  }
}

// ------------------------------------------------------------------
// residual add + LayerNorm (one block per row, D=1024, 4 elems/thread)
// ------------------------------------------------------------------
__global__ __launch_bounds__(256)
void k_add_ln(const float* __restrict__ x, const float* __restrict__ dl,
              const float* __restrict__ g, const float* __restrict__ bt,
              float* __restrict__ outF, __bf16* __restrict__ outB)
{
  __shared__ float red[2][8];
  const int tid = threadIdx.x, lane = tid & 31, wid = tid >> 5;
  const size_t row = blockIdx.x;
  const float* xr = x + row * D_;
  const float* dr = dl + row * D_;

  float v[4]; float s1 = 0.f, s2 = 0.f;
#pragma unroll
  for (int j = 0; j < 4; ++j) {
    int i = tid + j * 256;
    float t = xr[i] + dr[i];
    v[j] = t; s1 += t; s2 += t * t;
  }
#pragma unroll
  for (int x2 = 1; x2 < 32; x2 <<= 1) {
    s1 += __shfl_xor(s1, x2, 32);
    s2 += __shfl_xor(s2, x2, 32);
  }
  if (lane == 0) { red[0][wid] = s1; red[1][wid] = s2; }
  __syncthreads();
  s1 = 0.f; s2 = 0.f;
#pragma unroll
  for (int k = 0; k < 8; ++k) { s1 += red[0][k]; s2 += red[1][k]; }

  const float mean = s1 * (1.0f / D_);
  const float var  = s2 * (1.0f / D_) - mean * mean;
  const float rstd = rsqrtf(var + 1e-5f);
#pragma unroll
  for (int j = 0; j < 4; ++j) {
    int i = tid + j * 256;
    float y = (v[j] - mean) * rstd * g[i] + bt[i];
    if (outF) outF[row * D_ + i] = y;
    if (outB) outB[row * D_ + i] = (__bf16)y;
  }
}

// ------------------------------------------------------------------
// host orchestration
// ------------------------------------------------------------------
extern "C" void kernel_launch(void* const* d_in, const int* in_sizes, int n_in,
                              void* d_out, int out_size, void* d_ws, size_t ws_size,
                              hipStream_t stream)
{
  (void)in_sizes; (void)n_in; (void)out_size; (void)ws_size;

  const float* x      = (const float*)d_in[0];
  const float* enc    = (const float*)d_in[1];
  const float* sa_wq  = (const float*)d_in[4];
  const float* sa_bq  = (const float*)d_in[5];
  const float* sa_wk  = (const float*)d_in[6];
  const float* sa_bk  = (const float*)d_in[7];
  const float* sa_wv  = (const float*)d_in[8];
  const float* sa_bv  = (const float*)d_in[9];
  const float* sa_wo  = (const float*)d_in[10];
  const float* sa_bo  = (const float*)d_in[11];
  const float* ca_in_w  = (const float*)d_in[12];
  const float* ca_in_b  = (const float*)d_in[13];
  const float* ca_out_w = (const float*)d_in[14];
  const float* ca_out_b = (const float*)d_in[15];
  const float* ff_w1  = (const float*)d_in[16];
  const float* ff_b1  = (const float*)d_in[17];
  const float* ff_w2  = (const float*)d_in[18];
  const float* ff_b2  = (const float*)d_in[19];
  const float* n1_g = (const float*)d_in[20];
  const float* n1_b = (const float*)d_in[21];
  const float* n2_g = (const float*)d_in[22];
  const float* n2_b = (const float*)d_in[23];
  const float* n3_g = (const float*)d_in[24];
  const float* n3_b = (const float*)d_in[25];

  char* wsp = (char*)d_ws;
  auto alloc = [&](size_t bytes) -> void* {
    void* p = (void*)wsp;
    wsp += (bytes + 255) & ~(size_t)255;
    return p;
  };

  const size_t DD = (size_t)D_ * D_;
  const size_t FD = (size_t)DFF_ * D_;
  const size_t TD = (size_t)TOK * D_;
  const size_t TF = (size_t)TOK * DFF_;

  __bf16* wqb  = (__bf16*)alloc(DD * 2);
  __bf16* wkb  = (__bf16*)alloc(DD * 2);
  __bf16* wvb  = (__bf16*)alloc(DD * 2);
  __bf16* wob  = (__bf16*)alloc(DD * 2);
  __bf16* cwqb = (__bf16*)alloc(DD * 2);
  __bf16* cwkb = (__bf16*)alloc(DD * 2);
  __bf16* cwvb = (__bf16*)alloc(DD * 2);
  __bf16* cwob = (__bf16*)alloc(DD * 2);
  __bf16* fw1b = (__bf16*)alloc(FD * 2);
  __bf16* fw2b = (__bf16*)alloc(FD * 2);
  __bf16* xb   = (__bf16*)alloc(TD * 2);
  __bf16* encb = (__bf16*)alloc(TD * 2);
  __bf16* qb   = (__bf16*)alloc(TD * 2);
  __bf16* kbf  = (__bf16*)alloc(TD * 2);
  __bf16* vb   = (__bf16*)alloc(TD * 2);
  __bf16* vtb  = (__bf16*)alloc(TD * 2);
  __bf16* attb = (__bf16*)alloc(TD * 2);
  __bf16* x1b  = (__bf16*)alloc(TD * 2);
  __bf16* x2b  = (__bf16*)alloc(TD * 2);
  __bf16* ff1b = (__bf16*)alloc(TF * 2);
  float*  dF   = (float*)alloc(TD * 4);
  float*  x1F  = (float*)alloc(TD * 4);
  float*  x2F  = (float*)alloc(TD * 4);

  auto cvt = [&](const float* src, __bf16* dst, size_t n) {
    k_cvt_bf16<<<dim3((unsigned)(n / 256)), dim3(256), 0, stream>>>(src, dst, (int)n);
  };
  auto gemm = [&](const __bf16* A, const __bf16* W, const float* bias,
                  float* oF, __bf16* oB, int M, int N, int K, int relu) {
    k_gemm<<<dim3(N / 128, M / 128), dim3(256), 0, stream>>>(A, W, bias, oF, oB, M, N, K, relu);
  };

  // convert activations + weights to bf16
  cvt(x, xb, TD);
  cvt(enc, encb, TD);
  cvt(sa_wq, wqb, DD);
  cvt(sa_wk, wkb, DD);
  cvt(sa_wv, wvb, DD);
  cvt(sa_wo, wob, DD);
  cvt(ca_in_w, cwqb, DD);
  cvt(ca_in_w + DD, cwkb, DD);
  cvt(ca_in_w + 2 * DD, cwvb, DD);
  cvt(ca_out_w, cwob, DD);
  cvt(ff_w1, fw1b, FD);
  cvt(ff_w2, fw2b, FD);

  // ---- self attention ----
  gemm(xb, wqb, sa_bq, nullptr, qb,  TOK, D_, D_, 0);
  gemm(xb, wkb, sa_bk, nullptr, kbf, TOK, D_, D_, 0);
  gemm(xb, wvb, sa_bv, nullptr, vb,  TOK, D_, D_, 0);
  k_transpose_v<<<dim3((unsigned)(TD / 256)), dim3(256), 0, stream>>>(vb, vtb);
  k_attn<<<dim3(B_ * H_ * (S_ / 64)), dim3(128), 0, stream>>>(qb, kbf, vtb, attb, 1);
  gemm(attb, wob, sa_bo, dF, nullptr, TOK, D_, D_, 0);
  k_add_ln<<<dim3(TOK), dim3(256), 0, stream>>>(x, dF, n1_g, n1_b, x1F, x1b);

  // ---- cross attention ----
  gemm(x1b,  cwqb, ca_in_b,          nullptr, qb,  TOK, D_, D_, 0);
  gemm(encb, cwkb, ca_in_b + D_,     nullptr, kbf, TOK, D_, D_, 0);
  gemm(encb, cwvb, ca_in_b + 2 * D_, nullptr, vb,  TOK, D_, D_, 0);
  k_transpose_v<<<dim3((unsigned)(TD / 256)), dim3(256), 0, stream>>>(vb, vtb);
  k_attn<<<dim3(B_ * H_ * (S_ / 64)), dim3(128), 0, stream>>>(qb, kbf, vtb, attb, 0);
  gemm(attb, cwob, ca_out_b, dF, nullptr, TOK, D_, D_, 0);
  k_add_ln<<<dim3(TOK), dim3(256), 0, stream>>>(x1F, dF, n2_g, n2_b, x2F, x2b);

  // ---- feed forward ----
  gemm(x2b,  fw1b, ff_b1, nullptr, ff1b, TOK, DFF_, D_, 1);
  gemm(ff1b, fw2b, ff_b2, dF, nullptr,   TOK, D_, DFF_, 0);
  k_add_ln<<<dim3(TOK), dim3(256), 0, stream>>>(x2F, dF, n3_g, n3_b, (float*)d_out, nullptr);
}
